// LSTMWordChar_21715354649783
// MI455X (gfx1250) — matrix operand order repaired and checked by reading, not compiled, not run on hardware
//
#include <hip/hip_runtime.h>

// ---------------------------------------------------------------------------
// LSTM word/char tagger for MI455X (gfx1250), wave32 + WMMA bf16 + TDM.
// S=2048 words, CMAX=12 chars, E=512, HC=128, H=512, T=64.
// ---------------------------------------------------------------------------

#define S_LEN 2048
#define CMAXC 12
#define EDIM  512
#define HCH   128
#define HW    512
#define TCLS  64
#define GC    (4 * HCH)    // 512 char-gate rows
#define GW    (4 * HW)     // 2048 word-gate rows
#define KWDIM (EDIM + HCH) // 640
#define NBLK_WORD 16

typedef __bf16 bf16_t;
typedef bf16_t bf16x16 __attribute__((ext_vector_type(16)));
typedef float  f32x8   __attribute__((ext_vector_type(8)));
typedef unsigned int u32x4 __attribute__((ext_vector_type(4)));
typedef int i32x4 __attribute__((ext_vector_type(4)));
typedef int i32x8 __attribute__((ext_vector_type(8)));

#if defined(__has_builtin)
#if __has_builtin(__builtin_amdgcn_tensor_load_to_lds)
#define USE_TDM 1
#endif
#endif
#ifndef USE_TDM
#define USE_TDM 0
#endif

struct FragU { unsigned int r[8]; };

__device__ __forceinline__ bf16x16 as_bf16x16(const FragU &f) {
  return __builtin_bit_cast(bf16x16, f);
}

__device__ __forceinline__ f32x8 wmma_bf16(const FragU &a, const FragU &b, f32x8 c) {
  // (neg_a, A, neg_b, B, c_mod, C, reuse_a, reuse_b)
  return __builtin_amdgcn_wmma_f32_16x16x32_bf16(false, as_bf16x16(a),
                                                 false, as_bf16x16(b),
                                                 (short)0, c, false, false);
}

__device__ __forceinline__ unsigned short f2bf(float f) {
  unsigned int u = __float_as_uint(f);
  u += 0x7FFFu + ((u >> 16) & 1u);               // round-to-nearest-even
  return (unsigned short)(u >> 16);
}
__device__ __forceinline__ float bf2f(unsigned short h) {
  return __uint_as_float(((unsigned int)h) << 16);
}
__device__ __forceinline__ float sigf(float x) { return 1.0f / (1.0f + __expf(-x)); }

// A-fragment (16x32 bf16) K offset per ISA 7.12.2: lanes 0-15 hold K {0..7,16..23},
// lanes 16-31 hold K {8..15,24..31}; VGPR v holds a consecutive K pair.
__device__ __forceinline__ int a_koff(int v, int half) {
  return ((v < 4) ? 2 * v : 2 * v + 8) + 8 * half;
}

#if USE_TDM
// ---------------------------------------------------------------------------
// Tensor Data Mover: build a D# (ISA 8.3/8.4) and issue TENSOR_LOAD_TO_LDS.
// Up-to-3D tile of bf16 (data_size code 1 = 2 bytes), no padding/iterate.
// Address formula: addr = global + 2*(x + y*stride_d0 + z*stride_d1).
// ---------------------------------------------------------------------------
__device__ __forceinline__ void tdm_load_to_lds(
    void *lds_dst, const void *gsrc,
    unsigned tensor_d0, unsigned tensor_d1, unsigned tensor_d2,
    unsigned tile_d0, unsigned tile_d1, unsigned tile_d2,
    unsigned long long stride_d0, unsigned long long stride_d1) {
  const unsigned long long ga = (unsigned long long)gsrc;
  const unsigned lds = (unsigned)(unsigned long long)lds_dst; // LDS aperture low bits
  u32x4 g0;
  g0.x = 0x1u;                                        // count=1, user descriptor
  g0.y = lds;                                         // lds_addr
  g0.z = (unsigned)(ga & 0xFFFFFFFFu);                // global_addr[31:0]
  g0.w = (unsigned)((ga >> 32) & 0x01FFFFFFu) | (2u << 30); // addr[56:32] | type=2
  i32x8 g1;
  g1[0] = (int)(1u << 16);                            // wg_mask=0, data_size=2B
  g1[1] = (int)((tensor_d0 & 0xFFFFu) << 16);         // bits 48..63: dim0 lo
  g1[2] = (int)((tensor_d0 >> 16) | ((tensor_d1 & 0xFFFFu) << 16));
  g1[3] = (int)((tensor_d1 >> 16) | (tile_d0 << 16)); // dim1 hi | tile_dim0
  g1[4] = (int)(tile_d1 | (tile_d2 << 16));           // tile_dim1 | tile_dim2
  g1[5] = (int)(stride_d0 & 0xFFFFFFFFull);           // dim0_stride[31:0]
  g1[6] = (int)(((stride_d0 >> 32) & 0xFFFFull) |
                ((stride_d1 & 0xFFFFull) << 16));     // dim0_str[47:32]|dim1_str[15:0]
  g1[7] = (int)((stride_d1 >> 16) & 0xFFFFFFFFull);   // dim1_stride[47:16]
  i32x4 g2;
  g2[0] = (int)tensor_d2;                             // tensor_dim2
  g2[1] = 0;                                          // tensor_dim3
  g2[2] = 0;                                          // tensor_dim2_stride lo (3D tile unused)
  g2[3] = 0;                                          // dim2_stride hi | tile_dim3=0
  i32x4 g3 = {0, 0, 0, 0};
#if __clang_major__ >= 23
  i32x8 gx = {0, 0, 0, 0, 0, 0, 0, 0};
  __builtin_amdgcn_tensor_load_to_lds(g0, g1, g2, g3, gx, 0);
#else
  __builtin_amdgcn_tensor_load_to_lds(g0, g1, g2, g3, 0);
#endif
}
#endif // USE_TDM

// ---------------------------------------------------------------------------
// Prep: fp32 -> bf16 convert / transpose (B matrices stored K-major: B[k][n]=W[n][k])
// ---------------------------------------------------------------------------
__global__ void k_convert_bf16(unsigned short *__restrict__ dst,
                               const float *__restrict__ src, int n) {
  for (int i = blockIdx.x * blockDim.x + threadIdx.x; i < n;
       i += gridDim.x * blockDim.x)
    dst[i] = f2bf(src[i]);
}

__global__ void k_transpose_bf16(unsigned short *__restrict__ dst,
                                 const float *__restrict__ src, int R, int C) {
  int n = R * C;
  for (int i = blockIdx.x * blockDim.x + threadIdx.x; i < n;
       i += gridDim.x * blockDim.x) {
    int r = i / C, c = i - r * C;
    dst[(size_t)c * R + r] = f2bf(src[i]);
  }
}

__global__ void k_zero_u32(unsigned int *p, int n) {
  int i = blockIdx.x * blockDim.x + threadIdx.x;
  if (i < n) p[i] = 0u;
}

// ---------------------------------------------------------------------------
// pre_c[m, 0:512] = c_emb[chars[m]] @ W_ih_c^T + b_c   (m = s*12 + cpos)
// WMMA bf16, block = 8 waves -> 64x128 output tile; output stored bf16.
// ---------------------------------------------------------------------------
__global__ void k_pregemm_char(unsigned short *__restrict__ preC,
                               const unsigned short *__restrict__ cembB,
                               const unsigned short *__restrict__ BT, // [EDIM][GC]
                               const int *__restrict__ chars,
                               const float *__restrict__ bc) {
  const int lane = threadIdx.x & 31;
  const int wave = threadIdx.x >> 5;
  const int half = lane >> 4;
  const int l15  = lane & 15;
  const int row0 = blockIdx.y * 64 + (wave >> 1) * 16;
  const int col0 = blockIdx.x * 128 + (wave & 1) * 64;
  const int row  = row0 + l15;
  const int tok  = chars[row]; // flat (s*CMAX + cpos) == m
  const unsigned short *arow = cembB + (size_t)tok * EDIM;

  f32x8 acc[4];
#pragma unroll
  for (int j = 0; j < 4; ++j) acc[j] = (f32x8)(0.0f);

  for (int kb = 0; kb < EDIM; kb += 32) {
    FragU a;
#pragma unroll
    for (int v = 0; v < 8; ++v)
      a.r[v] = *(const unsigned int *)(arow + kb + a_koff(v, half));
#pragma unroll
    for (int j = 0; j < 4; ++j) {
      const int col = col0 + 16 * j + l15;
      FragU b;
#pragma unroll
      for (int v = 0; v < 8; ++v) {
        const int k0 = kb + a_koff(v, half);
        unsigned int lo = BT[(size_t)k0 * GC + col];
        unsigned int hi = BT[(size_t)(k0 + 1) * GC + col];
        b.r[v] = lo | (hi << 16);
      }
      acc[j] = wmma_bf16(a, b, acc[j]);
    }
  }
#pragma unroll
  for (int j = 0; j < 4; ++j) {
    const int col = col0 + 16 * j + l15;
    const float bias = bc[col];
#pragma unroll
    for (int r = 0; r < 8; ++r) {
      const int m = row0 + r + 8 * half;
      preC[(size_t)m * GC + col] = f2bf(acc[j][r] + bias);
    }
  }
}

// ---------------------------------------------------------------------------
// pre_w[s, 0:2048] = x_w[s] @ W_ih_w^T + b_w   (f32 output for the recurrence)
// ---------------------------------------------------------------------------
__global__ void k_pregemm_word(float *__restrict__ preW,
                               const unsigned short *__restrict__ xw, // [S][640] bf16
                               const unsigned short *__restrict__ BT, // [640][2048]
                               const float *__restrict__ bw) {
  const int lane = threadIdx.x & 31;
  const int wave = threadIdx.x >> 5;
  const int half = lane >> 4;
  const int l15  = lane & 15;
  const int row0 = blockIdx.y * 64 + (wave >> 1) * 16;
  const int col0 = blockIdx.x * 128 + (wave & 1) * 64;
  const unsigned short *arow = xw + (size_t)(row0 + l15) * KWDIM;

  f32x8 acc[4];
#pragma unroll
  for (int j = 0; j < 4; ++j) acc[j] = (f32x8)(0.0f);

  for (int kb = 0; kb < KWDIM; kb += 32) {
    FragU a;
#pragma unroll
    for (int v = 0; v < 8; ++v)
      a.r[v] = *(const unsigned int *)(arow + kb + a_koff(v, half));
#pragma unroll
    for (int j = 0; j < 4; ++j) {
      const int col = col0 + 16 * j + l15;
      FragU b;
#pragma unroll
      for (int v = 0; v < 8; ++v) {
        const int k0 = kb + a_koff(v, half);
        unsigned int lo = BT[(size_t)k0 * GW + col];
        unsigned int hi = BT[(size_t)(k0 + 1) * GW + col];
        b.r[v] = lo | (hi << 16);
      }
      acc[j] = wmma_bf16(a, b, acc[j]);
    }
  }
#pragma unroll
  for (int j = 0; j < 4; ++j) {
    const int col = col0 + 16 * j + l15;
    const float bias = bw[col];
#pragma unroll
    for (int r = 0; r < 8; ++r) {
      const int m = row0 + r + 8 * half;
      preW[(size_t)m * GW + col] = acc[j][r] + bias;
    }
  }
}

// ---------------------------------------------------------------------------
// Char LSTM: 32 blocks x 64 sentences. W_hh_c^T (128x512 bf16) resident in LDS
// (loaded by the Tensor Data Mover), h (bf16) / c (f32) state in LDS,
// per-step WMMA (64x128)x(128x512) + gates.
// Dynamic LDS: 128K (W) + 16K (h) + 32K (c) + 128K (z) = 304 KB (<320 KB/WGP).
// ---------------------------------------------------------------------------
__global__ void k_char_lstm(unsigned short *__restrict__ hcharB,
                            const unsigned short *__restrict__ preC,
                            const unsigned short *__restrict__ WT, // [HCH][GC] bf16
                            const int *__restrict__ wlen) {
  extern __shared__ unsigned char smem[];
  unsigned short *Wl = (unsigned short *)smem;               // 131072 B
  unsigned short *hl = (unsigned short *)(smem + 131072);    // 16384 B
  float *cl = (float *)(smem + 131072 + 16384);              // 32768 B
  float *zl = (float *)(smem + 131072 + 16384 + 32768);      // 131072 B

  const int tid  = threadIdx.x;
  const int lane = tid & 31;
  const int wave = tid >> 5;
  const int half = lane >> 4;
  const int l15  = lane & 15;
  const int s0   = blockIdx.x * 64;

#if USE_TDM
  // One DMA: 2D tile 512x128 bf16, contiguous rows -> LDS.
  if (tid < 32) {
    tdm_load_to_lds(Wl, WT, GC, HCH, 0, GC, HCH, 0,
                    (unsigned long long)GC, 0ull);
    __builtin_amdgcn_s_wait_tensorcnt(0);
  }
#else
  { // fallback: cooperative weights -> LDS
    const unsigned int *src = (const unsigned int *)WT;
    unsigned int *dst = (unsigned int *)Wl;
    for (int i = tid; i < HCH * GC / 2; i += blockDim.x) dst[i] = src[i];
  }
#endif
  for (int e = tid; e < 64 * HCH; e += blockDim.x) { cl[e] = 0.0f; hl[e] = 0; }
  __syncthreads();

  const int wm = wave & 3;   // 4 strips of 16 batch rows
  const int wn = wave >> 2;  // 2 halves of 256 gate cols
  const int arow = wm * 16 + l15;

  for (int t = 0; t < CMAXC; ++t) {
    f32x8 acc[16];
#pragma unroll
    for (int j = 0; j < 16; ++j) acc[j] = (f32x8)(0.0f);
#pragma unroll
    for (int kb = 0; kb < HCH; kb += 32) {
      FragU a;
#pragma unroll
      for (int v = 0; v < 8; ++v)
        a.r[v] = *(const unsigned int *)(hl + arow * HCH + kb + a_koff(v, half));
#pragma unroll
      for (int j = 0; j < 16; ++j) {
        const int col = wn * 256 + j * 16 + l15;
        FragU b;
#pragma unroll
        for (int v = 0; v < 8; ++v) {
          const int k0 = kb + a_koff(v, half);
          unsigned int lo = Wl[k0 * GC + col];
          unsigned int hi = Wl[(k0 + 1) * GC + col];
          b.r[v] = lo | (hi << 16);
        }
        acc[j] = wmma_bf16(a, b, acc[j]);
      }
    }
    // z = h@Whh^T + pre_c[:, t]  -> zl
#pragma unroll
    for (int j = 0; j < 16; ++j) {
      const int col = wn * 256 + j * 16 + l15;
#pragma unroll
      for (int r = 0; r < 8; ++r) {
        const int m = wm * 16 + r + 8 * half;
        const int srow = s0 + m;
        float pre = bf2f(preC[((size_t)srow * CMAXC + t) * GC + col]);
        zl[m * GC + col] = acc[j][r] + pre;
      }
    }
    __syncthreads();
    // gates + masked state update (mask: t < word_len[s])
    for (int e = tid; e < 64 * HCH; e += blockDim.x) {
      const int s = e >> 7, j = e & 127;
      const float zi = zl[s * GC + j];
      const float zf = zl[s * GC + HCH + j];
      const float zg = zl[s * GC + 2 * HCH + j];
      const float zo = zl[s * GC + 3 * HCH + j];
      const float cp = cl[e];
      const float c2 = sigf(zf) * cp + sigf(zi) * tanhf(zg);
      const float h2 = sigf(zo) * tanhf(c2);
      if (t < wlen[s0 + s]) { cl[e] = c2; hl[e] = f2bf(h2); }
    }
    __syncthreads();
  }
  for (int e = tid; e < 64 * HCH; e += blockDim.x) {
    const int s = e >> 7, j = e & 127;
    hcharB[(size_t)(s0 + s) * HCH + j] = hl[e];
  }
}

// ---------------------------------------------------------------------------
// x_w = concat(w_emb[sentence] (bf16), h_char (bf16))  -> [S][640]
// ---------------------------------------------------------------------------
__global__ void k_build_xw(unsigned short *__restrict__ xw,
                           const float *__restrict__ wemb,
                           const int *__restrict__ sent,
                           const unsigned short *__restrict__ hcharB) {
  const int total = S_LEN * KWDIM;
  for (int idx = blockIdx.x * blockDim.x + threadIdx.x; idx < total;
       idx += gridDim.x * blockDim.x) {
    const int s = idx / KWDIM, k = idx - s * KWDIM;
    unsigned short v;
    if (k < EDIM) v = f2bf(wemb[(size_t)sent[s] * EDIM + k]);
    else          v = hcharB[s * HCH + (k - EDIM)];
    xw[idx] = v;
  }
}

// ---------------------------------------------------------------------------
// Word LSTM: 16 resident workgroups, each owns 32 h-elements + the 128
// matching gate rows of W_hh_w in LDS (bf16, 128 KB). The TDM fetches the
// strided 4x32-row gather as one 3D tile. Per step: 2 KB of h from L2, 256
// half-dot products, gate math, then a device-scope monotone atomic barrier
// chains the 2048 sequential steps.
// ---------------------------------------------------------------------------
__global__ void k_word_lstm(float *__restrict__ hAll,
                            const float *__restrict__ preW,
                            const unsigned short *__restrict__ Whh, // [GW][HW] bf16
                            unsigned int *__restrict__ bar) {
  extern __shared__ unsigned char smem[];
  unsigned short *Wl = (unsigned short *)smem;                       // 131072 B
  float *hlds  = (float *)(smem + 131072);                           // 2048 B
  float *zpart = (float *)(smem + 131072 + 2048);                    // 1024 B
  float *zrow  = (float *)(smem + 131072 + 2048 + 1024);             // 512 B
  float *cst   = (float *)(smem + 131072 + 2048 + 1024 + 512);       // 128 B

  const int tid = threadIdx.x;
  const int b = blockIdx.x;
  const int rl = tid >> 1;        // 0..127: local gate row
  const int halfk = tid & 1;      // which 256-wide half of K
  const int gate = rl >> 5, gi = rl & 31;
  const int nrow = gate * HW + b * 32 + gi; // global gate row in [0, 2048)

#if USE_TDM
  // 3D tile: x = 512 K-elems, y = 32 rows (stride 512), z = 4 gate blocks
  // (stride 512 rows * 512 = 262144 elems). Lands in LDS as rl = gate*32+gi.
  if (tid < 32) {
    tdm_load_to_lds(Wl, Whh + (size_t)b * 32 * HW,
                    HW, GW, 4,            // tensor dims (OOB bounds)
                    HW, 32, 4,            // tile dims
                    (unsigned long long)HW,
                    (unsigned long long)(4 * HW) * HW / 4 * 2 / 2); // 512*512
    __builtin_amdgcn_s_wait_tensorcnt(0);
  }
#else
  { // fallback: gather this block's 128 W_hh_w rows into LDS (bf16 pairs)
    const unsigned int *src = (const unsigned int *)Whh;
    unsigned int *dst = (unsigned int *)Wl;
    for (int i = tid; i < 128 * (HW / 2); i += blockDim.x) {
      const int r = i / (HW / 2), c = i - r * (HW / 2);
      const int gr = (r >> 5) * HW + b * 32 + (r & 31);
      dst[r * (HW / 2) + c] = src[(size_t)gr * (HW / 2) + c];
    }
  }
#endif
  if (tid < 32) cst[tid] = 0.0f;
  __syncthreads();

  const unsigned int *wr32 = (const unsigned int *)Wl + rl * (HW / 2) + halfk * 128;

  for (int t = 0; t < S_LEN; ++t) {
    __builtin_prefetch(&preW[(size_t)(t + 1) * GW + nrow], 0, 0);
    float sum = 0.0f;
    if (t > 0) {
      for (int k = tid; k < HW; k += blockDim.x)
        hlds[k] = hAll[(size_t)(t - 1) * HW + k];
      __syncthreads();
      const float *hp = hlds + halfk * 256;
#pragma unroll 8
      for (int k2 = 0; k2 < 128; ++k2) {
        const unsigned int w2 = wr32[k2];
        sum += bf2f((unsigned short)w2) * hp[2 * k2];
        sum += bf2f((unsigned short)(w2 >> 16)) * hp[2 * k2 + 1];
      }
    }
    zpart[tid] = sum;
    __syncthreads();
    if (halfk == 0)
      zrow[rl] = zpart[tid] + zpart[tid + 1] + preW[(size_t)t * GW + nrow];
    __syncthreads();
    if (tid < 32) {
      const float zi = zrow[tid], zf = zrow[32 + tid];
      const float zg = zrow[64 + tid], zo = zrow[96 + tid];
      float c = cst[tid];
      c = sigf(zf) * c + sigf(zi) * tanhf(zg);
      const float h = sigf(zo) * tanhf(c);
      cst[tid] = c;
      hAll[(size_t)t * HW + b * 32 + tid] = h;
    }
    __threadfence();           // release our h slice device-wide
    __syncthreads();
    if (tid == 0) {
      atomicAdd(bar, 1u);
      while (__hip_atomic_load(bar, __ATOMIC_ACQUIRE,
                               __HIP_MEMORY_SCOPE_AGENT) <
             (unsigned)(NBLK_WORD * (t + 1))) { }
    }
    __syncthreads();
  }
}

// ---------------------------------------------------------------------------
// tag_space = h_all @ W_lin^T + b_lin, then log_softmax over T=64.
// ---------------------------------------------------------------------------
__global__ void k_tagger(float *__restrict__ out, const float *__restrict__ hAll,
                         const float *__restrict__ Wlin,
                         const float *__restrict__ blin) {
  __shared__ float wred[2];
  __shared__ float wsum[2];
  const int s = blockIdx.x, j = threadIdx.x;
  const int lane = j & 31, wv = j >> 5;
  const float *hp = hAll + (size_t)s * HW;
  const float *wp = Wlin + (size_t)j * HW;
  float sum = blin[j];
#pragma unroll 4
  for (int k = 0; k < HW; ++k) sum += hp[k] * wp[k];

  float m = sum;
  for (int off = 16; off > 0; off >>= 1) m = fmaxf(m, __shfl_xor(m, off, 32));
  if (lane == 0) wred[wv] = m;
  __syncthreads();
  const float gm = fmaxf(wred[0], wred[1]);
  float e = __expf(sum - gm);
  for (int off = 16; off > 0; off >>= 1) e += __shfl_xor(e, off, 32);
  if (lane == 0) wsum[wv] = e;
  __syncthreads();
  const float gs = wsum[0] + wsum[1];
  out[(size_t)s * TCLS + j] = sum - gm - logf(gs);
}

// ---------------------------------------------------------------------------
extern "C" void kernel_launch(void *const *d_in, const int *in_sizes, int n_in,
                              void *d_out, int out_size, void *d_ws,
                              size_t ws_size, hipStream_t stream) {
  const int *sentence = (const int *)d_in[0];
  const int *chars    = (const int *)d_in[1];
  const int *wlen     = (const int *)d_in[2];
  const float *wemb   = (const float *)d_in[3];
  const float *cemb   = (const float *)d_in[4];
  const float *Wihc   = (const float *)d_in[5];
  const float *Whhc   = (const float *)d_in[6];
  const float *bc     = (const float *)d_in[7];
  const float *Wihw   = (const float *)d_in[8];
  const float *Whhw   = (const float *)d_in[9];
  const float *bw     = (const float *)d_in[10];
  const float *Wlin   = (const float *)d_in[11];
  const float *blin   = (const float *)d_in[12];
  float *out = (float *)d_out;

  unsigned char *ws = (unsigned char *)d_ws;
  size_t off = 0;
  auto alloc = [&](size_t bytes) {
    unsigned char *p = ws + off;
    off += (bytes + 255) & ~(size_t)255;
    return p;
  };
  unsigned short *cembB  = (unsigned short *)alloc((size_t)26 * EDIM * 2);
  unsigned short *WcihT  = (unsigned short *)alloc((size_t)EDIM * GC * 2);
  unsigned short *WchhT  = (unsigned short *)alloc((size_t)HCH * GC * 2);
  unsigned short *WwihT  = (unsigned short *)alloc((size_t)KWDIM * GW * 2);
  unsigned short *WwhhB  = (unsigned short *)alloc((size_t)GW * HW * 2);
  unsigned short *preC   = (unsigned short *)alloc((size_t)S_LEN * CMAXC * GC * 2);
  unsigned short *hcharB = (unsigned short *)alloc((size_t)S_LEN * HCH * 2);
  unsigned short *xw     = (unsigned short *)alloc((size_t)S_LEN * KWDIM * 2);
  float *preW = (float *)alloc((size_t)S_LEN * GW * 4);
  float *hAll = (float *)alloc((size_t)S_LEN * HW * 4);
  unsigned int *bar = (unsigned int *)alloc(256);

  // weight prep (fp32 -> bf16; B matrices K-major)
  k_convert_bf16<<<64, 256, 0, stream>>>(cembB, cemb, 26 * EDIM);
  k_transpose_bf16<<<512, 256, 0, stream>>>(WcihT, Wihc, GC, EDIM);
  k_transpose_bf16<<<256, 256, 0, stream>>>(WchhT, Whhc, GC, HCH);
  k_transpose_bf16<<<2048, 256, 0, stream>>>(WwihT, Wihw, GW, KWDIM);
  k_convert_bf16<<<2048, 256, 0, stream>>>(WwhhB, Whhw, GW * HW);

  // pre_c (fused char-embedding gather + WMMA GEMM)
  k_pregemm_char<<<dim3(GC / 128, (S_LEN * CMAXC) / 64), 256, 0, stream>>>(
      preC, cembB, WcihT, chars, bc);

  // char LSTM (TDM-loaded LDS weights, WMMA recurrence)
  const size_t smem_char = 131072 + 16384 + 32768 + 131072; // 304 KB
  k_char_lstm<<<S_LEN / 64, 256, smem_char, stream>>>(hcharB, preC, WchhT, wlen);

  // x_w concat, then pre_w WMMA GEMM
  k_build_xw<<<1024, 256, 0, stream>>>(xw, wemb, sentence, hcharB);
  k_pregemm_word<<<dim3(GW / 128, S_LEN / 64), 256, 0, stream>>>(preW, xw,
                                                                 WwihT, bw);

  // word LSTM (16 resident WGs + atomic chain barrier)
  k_zero_u32<<<1, 64, 0, stream>>>(bar, 64);
  const size_t smem_word = 131072 + 2048 + 1024 + 512 + 128;
  k_word_lstm<<<NBLK_WORD, 256, smem_word, stream>>>(hAll, preW, WwhhB, bar);

  // classifier + log-softmax
  k_tagger<<<S_LEN, TCLS, 0, stream>>>(out, hAll, Wlin, blin);

  (void)in_sizes; (void)n_in; (void)out_size; (void)ws_size;
}
